// MultiHeadAttention_15599321219189
// MI455X (gfx1250) — matrix operand order repaired and checked by reading, not compiled, not run on hardware
//
#include <hip/hip_runtime.h>
#include <hip/hip_bf16.h>

// Problem constants (from reference)
#define BS 4
#define LQ 2048
#define LKV 2048
#define D_MODEL 512
#define HEADS 8
#define DKH 64           // D_MODEL / HEADS
#define MROWS (BS * LQ)  // 8192

typedef __attribute__((ext_vector_type(16))) _Float16 v16h;
typedef __attribute__((ext_vector_type(8)))  float    v8f;

static __device__ inline v8f vzero8() {
  v8f v;
#pragma unroll
  for (int i = 0; i < 8; ++i) v[i] = 0.0f;
  return v;
}

static __device__ inline v8f wmma16x16x32(v16h a, v16h b, v8f c) {
  return __builtin_amdgcn_wmma_f32_16x16x32_f16(
      false, a, false, b, (short)0, c, false, false);
}

// 16 contiguous halves (32B): B-fragment (Bt[N][K]; lanes 0-15 K=0..15,
// lanes 16-31 K=16..31 via caller's +16-half offset).
static __device__ inline v16h ld16_contig(const _Float16* p) {
  union { v16h v; uint4 u[2]; } r;
  r.u[0] = *(const uint4*)p;
  r.u[1] = *(const uint4*)(p + 8);
  return r.v;
}

// A-fragment: halves 0..7 at p, halves 8..15 at p+16 (K and K+16 groups).
static __device__ inline v16h ld16_a(const _Float16* p) {
  union { v16h v; uint4 u[2]; } r;
  r.u[0] = *(const uint4*)p;
  r.u[1] = *(const uint4*)(p + 16);
  return r.v;
}

// ---- CDNA5 async global->LDS DMA (ASYNCcnt-tracked, bypasses VGPRs) ----
// LDS offset for a __shared__ object is the low 32 bits of its generic addr.
static __device__ inline void async_cp_b128(void* lds_dst, const _Float16* g) {
  unsigned int loff = (unsigned int)(size_t)lds_dst;
  asm volatile("global_load_async_to_lds_b128 %0, %1, off"
               :
               : "v"(loff), "v"(g)
               : "memory");
}
static __device__ inline void wait_async0() {
  asm volatile("s_wait_asynccnt 0x0" ::: "memory");
}

// ---------------------------------------------------------------------------
// fp32 -> fp16 conversion (activations, x4 vectorized)
// ---------------------------------------------------------------------------
__global__ __launch_bounds__(256) void k_f32_to_f16(const float* __restrict__ x,
                                                    _Float16* __restrict__ y,
                                                    int n) {
  int i = (blockIdx.x * blockDim.x + threadIdx.x) * 4;
  if (i + 3 < n) {
    float4 v = *(const float4*)(x + i);
    y[i + 0] = (_Float16)v.x;
    y[i + 1] = (_Float16)v.y;
    y[i + 2] = (_Float16)v.z;
    y[i + 3] = (_Float16)v.w;
  } else {
    for (; i < n; ++i) y[i] = (_Float16)x[i];
  }
}

// fp32 [K,N] -> fp16 transposed [N,K]  (512x512 weight matrices)
__global__ __launch_bounds__(256) void k_wT_f16(const float* __restrict__ w,
                                                _Float16* __restrict__ wt) {
  int idx = blockIdx.x * 256 + threadIdx.x;  // = k*512 + n (coalesced read)
  int k = idx >> 9, n = idx & 511;
  wt[(size_t)n * 512 + k] = (_Float16)w[idx];
}

// ---------------------------------------------------------------------------
// WMMA GEMM: C[M,N] = A[M,K](f16) * Bt[N,K](f16, pre-transposed) + bias
// 128x128x32 tiles, 8 waves x (64x32), double-buffered LDS filled by
// async global->LDS b128 DMA.  mode: 0 = f16 row-major, 1 = f32 row-major,
// 2 = f16 transposed-per-batch ([b][col][row%LKV]) for the V projection.
// ---------------------------------------------------------------------------
#define GBM 128
#define GBN 128
#define GBK 32
#define LDT 40  // padded LDS row stride (80B, 16B aligned)

__global__ __launch_bounds__(256) void k_gemm_wmma(
    const _Float16* __restrict__ A, const _Float16* __restrict__ Bt,
    const float* __restrict__ bias, _Float16* __restrict__ Ch,
    float* __restrict__ Cf, int M, int N, int K, int mode) {
  __shared__ _Float16 As[2][GBM * LDT];
  __shared__ _Float16 Bs[2][GBN * LDT];  // Bs[n][k]

  const int t    = threadIdx.x;
  const int lane = t & 31;
  const int wave = t >> 5;
  const int r    = lane & 15;
  const int hi   = lane >> 4;

  const int m0 = blockIdx.y * GBM;
  const int n0 = blockIdx.x * GBN;
  const int wm = (wave >> 2) * 64;  // 0 or 64
  const int wn = (wave & 3) * 32;   // 0,32,64,96

  // staging assignment: thread covers 16 halves of one row of each tile
  const int srow = t >> 1;
  const int sseg = (t & 1) * 16;
  const _Float16* gA = A  + (size_t)(m0 + srow) * K + sseg;
  const _Float16* gB = Bt + (size_t)(n0 + srow) * K + sseg;

  v8f acc[4][2];
#pragma unroll
  for (int mi = 0; mi < 4; ++mi)
#pragma unroll
    for (int ni = 0; ni < 2; ++ni) acc[mi][ni] = vzero8();

  const int nk = K / GBK;
  // prologue: stage tile 0 into buffer 0
  {
    async_cp_b128(&As[0][srow * LDT + sseg],     gA);
    async_cp_b128(&As[0][srow * LDT + sseg + 8], gA + 8);
    async_cp_b128(&Bs[0][srow * LDT + sseg],     gB);
    async_cp_b128(&Bs[0][srow * LDT + sseg + 8], gB + 8);
  }

  for (int it = 0; it < nk; ++it) {
    const int cur = it & 1;
    wait_async0();
    __syncthreads();  // tile `cur` resident & visible
    if (it + 1 < nk) {
      const int nxt = cur ^ 1;
      const int kk  = (it + 1) * GBK;
      async_cp_b128(&As[nxt][srow * LDT + sseg],     gA + kk);
      async_cp_b128(&As[nxt][srow * LDT + sseg + 8], gA + kk + 8);
      async_cp_b128(&Bs[nxt][srow * LDT + sseg],     gB + kk);
      async_cp_b128(&Bs[nxt][srow * LDT + sseg + 8], gB + kk + 8);
    }
    v16h afr[4];
#pragma unroll
    for (int mi = 0; mi < 4; ++mi)
      afr[mi] = ld16_contig(&As[cur][(wm + mi * 16 + r) * LDT + hi * 16]);
    v16h bfr[2];
#pragma unroll
    for (int ni = 0; ni < 2; ++ni)
      bfr[ni] = ld16_contig(&Bs[cur][(wn + ni * 16 + r) * LDT + hi * 16]);
#pragma unroll
    for (int mi = 0; mi < 4; ++mi)
#pragma unroll
      for (int ni = 0; ni < 2; ++ni)
        acc[mi][ni] = wmma16x16x32(afr[mi], bfr[ni], acc[mi][ni]);
    __syncthreads();  // all waves done with `cur` before it is restaged
  }

  // epilogue (C layout: element (j,lane) -> row j+8*hi, col lane&15)
#pragma unroll
  for (int mi = 0; mi < 4; ++mi)
#pragma unroll
    for (int ni = 0; ni < 2; ++ni) {
      int col  = n0 + wn + ni * 16 + r;
      float bv = bias ? bias[col] : 0.0f;
#pragma unroll
      for (int j = 0; j < 8; ++j) {
        int row   = m0 + wm + mi * 16 + j + 8 * hi;
        float val = acc[mi][ni][j] + bv;
        if (mode == 0) {
          Ch[(size_t)row * N + col] = (_Float16)val;
        } else if (mode == 1) {
          Cf[(size_t)row * N + col] = val;
        } else {  // V projection: [b][col][row%LKV]
          int b  = row >> 11;          // row / LKV
          int rb = row & (LKV - 1);
          Ch[((size_t)b * N + col) * LKV + rb] = (_Float16)val;
        }
      }
    }
}

// ---------------------------------------------------------------------------
// Flash attention with log-weight bias:  wp = softmax(s + log w).
// 8 waves x 16 queries; KV streamed in 64-blocks, double-buffered LDS via
// async DMA.  K tile is row-major (already the B^T layout for Q·K^T);
// V comes pre-transposed from the projection ([b][d][kv]).
// ---------------------------------------------------------------------------
#define AQ  128
#define AKV 64
#define LDKS 72  // padded LDS stride (144B, 16B aligned)

__global__ __launch_bounds__(256) void k_attn_wmma(
    const _Float16* __restrict__ Q, const _Float16* __restrict__ Kh,
    const _Float16* __restrict__ VT, const float* __restrict__ wts,
    _Float16* __restrict__ O) {
  __shared__ _Float16 Ks[2][AKV * LDKS];    // [kv][d]
  __shared__ _Float16 Vt[2][DKH * LDKS];    // [d][kv]
  __shared__ _Float16 Ps[8 * 16 * LDKS];    // per-wave P staging [16][64]

  const int t    = threadIdx.x;
  const int lane = t & 31;
  const int wave = t >> 5;
  const int r    = lane & 15;
  const int hi   = lane >> 4;

  int bid = blockIdx.x;
  const int qt = bid % (LQ / AQ); bid /= (LQ / AQ);
  const int h  = bid % HEADS;
  const int b  = bid / HEADS;

  const int    q0   = qt * AQ + wave * 16;
  const size_t rowQ = (size_t)b * LQ + q0;
  const int    colH = h * DKH;

  // staging assignment: row = t>>2 in [0,64), seg = (t&3)*16
  const int srow = t >> 2;
  const int sseg = (t & 3) * 16;
  const _Float16* gK = Kh + ((size_t)b * LKV + srow) * D_MODEL + colH + sseg;
  const _Float16* gV = VT + ((size_t)b * D_MODEL + colH + srow) * LKV + sseg;

  // Q fragments for both K-substeps, loaded once from global.
  v16h qf[2];
  {
    const _Float16* qp = Q + (rowQ + r) * D_MODEL + colH + hi * 16;
    qf[0] = ld16_contig(qp);
    qf[1] = ld16_contig(qp + 32);
  }

  float rm[8], rs[8];
#pragma unroll
  for (int j = 0; j < 8; ++j) { rm[j] = -1e30f; rs[j] = 0.0f; }
  v8f oacc[4];
#pragma unroll
  for (int dt = 0; dt < 4; ++dt) oacc[dt] = vzero8();

  const float iscale = 0.125f;  // 1/sqrt(64)
  _Float16* Pw = &Ps[wave * 16 * LDKS];

  const int nit = LKV / AKV;
  // prologue: stage KV block 0
  {
    async_cp_b128(&Ks[0][srow * LDKS + sseg],     gK);
    async_cp_b128(&Ks[0][srow * LDKS + sseg + 8], gK + 8);
    async_cp_b128(&Vt[0][srow * LDKS + sseg],     gV);
    async_cp_b128(&Vt[0][srow * LDKS + sseg + 8], gV + 8);
  }

  for (int it = 0; it < nit; ++it) {
    const int cur = it & 1;
    const int kv0 = it * AKV;
    wait_async0();
    __syncthreads();
    if (it + 1 < nit) {
      const int nxt = cur ^ 1;
      const int kn  = kv0 + AKV;
      async_cp_b128(&Ks[nxt][srow * LDKS + sseg],     gK + (size_t)kn * D_MODEL);
      async_cp_b128(&Ks[nxt][srow * LDKS + sseg + 8], gK + (size_t)kn * D_MODEL + 8);
      async_cp_b128(&Vt[nxt][srow * LDKS + sseg],     gV + kn);
      async_cp_b128(&Vt[nxt][srow * LDKS + sseg + 8], gV + kn + 8);
    }

    // ---- scores S = Q K^T * iscale + log(w), 4 column tiles of 16 ----
    float mnew[8];
#pragma unroll
    for (int j = 0; j < 8; ++j) mnew[j] = rm[j];
    v8f sc[4];
#pragma unroll
    for (int nt = 0; nt < 4; ++nt) {
      v8f a = vzero8();
#pragma unroll
      for (int ks = 0; ks < 2; ++ks) {
        v16h bf =
            ld16_contig(&Ks[cur][(nt * 16 + r) * LDKS + ks * 32 + hi * 16]);
        a = wmma16x16x32(qf[ks], bf, a);
      }
      float lw = __logf(wts[(size_t)b * LKV + kv0 + nt * 16 + r]);
#pragma unroll
      for (int j = 0; j < 8; ++j) {
        float v = a[j] * iscale + lw;
        a[j] = v;
        v = fmaxf(v, __shfl_xor(v, 1, 32));
        v = fmaxf(v, __shfl_xor(v, 2, 32));
        v = fmaxf(v, __shfl_xor(v, 4, 32));
        v = fmaxf(v, __shfl_xor(v, 8, 32));
        mnew[j] = fmaxf(mnew[j], v);
      }
      sc[nt] = a;
    }

    // ---- online softmax rescale ----
    float esc[8];
#pragma unroll
    for (int j = 0; j < 8; ++j) {
      esc[j] = __expf(rm[j] - mnew[j]);
      rm[j]  = mnew[j];
      rs[j] *= esc[j];
    }
#pragma unroll
    for (int dt = 0; dt < 4; ++dt)
#pragma unroll
      for (int j = 0; j < 8; ++j) oacc[dt][j] *= esc[j];

    // ---- P = exp(S-m); accumulate row sums; stage P (wave-local) ----
#pragma unroll
    for (int nt = 0; nt < 4; ++nt) {
#pragma unroll
      for (int j = 0; j < 8; ++j) {
        float p = __expf(sc[nt][j] - mnew[j]);
        float s = p;
        s += __shfl_xor(s, 1, 32);
        s += __shfl_xor(s, 2, 32);
        s += __shfl_xor(s, 4, 32);
        s += __shfl_xor(s, 8, 32);
        rs[j] += s;
        Pw[(j + 8 * hi) * LDKS + nt * 16 + r] = (_Float16)p;
      }
    }

    // ---- O += P @ V ----
#pragma unroll
    for (int ks = 0; ks < 2; ++ks) {
      v16h pf = ld16_contig(&Pw[r * LDKS + ks * 32 + hi * 16]);
#pragma unroll
      for (int dt = 0; dt < 4; ++dt) {
        v16h vf =
            ld16_contig(&Vt[cur][(dt * 16 + r) * LDKS + ks * 32 + hi * 16]);
        oacc[dt] = wmma16x16x32(pf, vf, oacc[dt]);
      }
    }
    __syncthreads();
  }

  // ---- normalize & write head slice of concat buffer ----
#pragma unroll
  for (int dt = 0; dt < 4; ++dt) {
    int col = colH + dt * 16 + r;
#pragma unroll
    for (int j = 0; j < 8; ++j) {
      size_t row = rowQ + j + 8 * hi;
      float inv  = 1.0f / rs[j];
      O[row * D_MODEL + col] = (_Float16)(oacc[dt][j] * inv);
    }
  }
}

// ---------------------------------------------------------------------------
// Host-side launch
// ---------------------------------------------------------------------------
extern "C" void kernel_launch(void* const* d_in, const int* in_sizes, int n_in,
                              void* d_out, int out_size, void* d_ws,
                              size_t ws_size, hipStream_t stream) {
  (void)in_sizes; (void)n_in; (void)out_size; (void)ws_size;
  const float* q   = (const float*)d_in[0];
  const float* k   = (const float*)d_in[1];
  const float* v   = (const float*)d_in[2];
  const float* wts = (const float*)d_in[3];
  const float* Wq  = (const float*)d_in[4];
  const float* bq  = (const float*)d_in[5];
  const float* Wk  = (const float*)d_in[6];
  const float* bk  = (const float*)d_in[7];
  const float* Wv  = (const float*)d_in[8];
  const float* bv  = (const float*)d_in[9];
  const float* Wo  = (const float*)d_in[10];
  const float* bo  = (const float*)d_in[11];
  float* out = (float*)d_out;

  const size_t actN = (size_t)MROWS * D_MODEL;   // 8192*512
  const size_t wN   = (size_t)D_MODEL * D_MODEL; // 512*512
  _Float16* p    = (_Float16*)d_ws;
  _Float16* q16  = p; p += actN;
  _Float16* k16  = p; p += actN;
  _Float16* v16  = p; p += actN;
  _Float16* WqT  = p; p += wN;   // pre-transposed f16 weights [N][K]
  _Float16* WkT  = p; p += wN;
  _Float16* WvT  = p; p += wN;
  _Float16* WoT  = p; p += wN;
  _Float16* Qp   = p; p += actN; // projected Q [8192,512]
  _Float16* Kp   = p; p += actN; // projected K [8192,512]
  _Float16* VpT  = p; p += actN; // projected V transposed [b][512][2048]
  _Float16* AO   = p; p += actN; // attention output (concat layout)

  // 1) convert activations; transpose-convert weights
  {
    int n = (int)actN, g = (n / 4 + 255) / 256;
    k_f32_to_f16<<<g, 256, 0, stream>>>(q, q16, n);
    k_f32_to_f16<<<g, 256, 0, stream>>>(k, k16, n);
    k_f32_to_f16<<<g, 256, 0, stream>>>(v, v16, n);
    int gw = (int)(wN / 256);
    k_wT_f16<<<gw, 256, 0, stream>>>(Wq, WqT);
    k_wT_f16<<<gw, 256, 0, stream>>>(Wk, WkT);
    k_wT_f16<<<gw, 256, 0, stream>>>(Wv, WvT);
    k_wT_f16<<<gw, 256, 0, stream>>>(Wo, WoT);
  }

  // 2) projections
  dim3 gg(D_MODEL / GBN, MROWS / GBM);  // (4, 64)
  k_gemm_wmma<<<gg, 256, 0, stream>>>(q16, WqT, bq, Qp, nullptr,
                                      MROWS, D_MODEL, D_MODEL, 0);
  k_gemm_wmma<<<gg, 256, 0, stream>>>(k16, WkT, bk, Kp, nullptr,
                                      MROWS, D_MODEL, D_MODEL, 0);
  k_gemm_wmma<<<gg, 256, 0, stream>>>(v16, WvT, bv, VpT, nullptr,
                                      MROWS, D_MODEL, D_MODEL, 2);

  // 3) attention
  int nblk = BS * HEADS * (LQ / AQ);  // 512
  k_attn_wmma<<<nblk, 256, 0, stream>>>(Qp, Kp, VpT, wts, AO);

  // 4) output projection (fp32)
  k_gemm_wmma<<<gg, 256, 0, stream>>>(AO, WoT, bo, nullptr, out,
                                      MROWS, D_MODEL, D_MODEL, 1);
}
